// _EncoderLayer_12257836663306
// MI455X (gfx1250) — matrix operand order repaired and checked by reading, not compiled, not run on hardware
//
#include <hip/hip_runtime.h>

// ---------------------------------------------------------------------------
// Types
// ---------------------------------------------------------------------------
typedef __bf16 bf16;
typedef __attribute__((ext_vector_type(4)))  __bf16 bf16x4;
typedef __attribute__((ext_vector_type(8)))  __bf16 bf16x8;
typedef __attribute__((ext_vector_type(16))) __bf16 bf16x16;
typedef __attribute__((ext_vector_type(8)))  float  f32x8;
typedef __attribute__((ext_vector_type(4)))  int    i32x4;

#define AS1 __attribute__((address_space(1)))
#define AS3 __attribute__((address_space(3)))

#if defined(__has_builtin)
#if __has_builtin(__builtin_amdgcn_global_load_async_to_lds_b128) && \
    __has_builtin(__builtin_amdgcn_s_wait_asynccnt)
#define USE_ASYNC_LDS 1
#endif
#endif
#ifndef USE_ASYNC_LDS
#define USE_ASYNC_LDS 0
#endif

#if USE_ASYNC_LDS
// builtin expects (global int4*, lds int4*, imm offset, imm cpol)
#define ASYNC_CP16(gsrc, ldst)                                            \
  __builtin_amdgcn_global_load_async_to_lds_b128(                         \
      (AS1 i32x4*)(unsigned long long)(const void*)(gsrc),                \
      (AS3 i32x4*)(unsigned long long)(void*)(ldst), 0, 0)
#endif

// ---------------------------------------------------------------------------
// Problem constants
// ---------------------------------------------------------------------------
#define NB   4        // batch
#define SS   1024     // sequence
#define DD   1024     // model dim
#define HH   16       // heads
#define DKK  64       // head dim
#define DFF  4096     // ffn dim
#define MT   4096     // NB*SS rows

#define WG   256      // threads per workgroup (8 waves of 32)
#define BMT  128      // M tile per workgroup (16 rows per wave)
#define BNT  64       // N tile per workgroup
#define KCT  32       // K step (one bf16 WMMA)
#define ASTR 40       // LDS row stride (bf16 elems), 16B aligned, pad 8
#define KSTR 72       // LDS row stride for 64-wide rows, pad 8

enum { ACT_NONE = 0, ACT_RELU = 1 };

__device__ __forceinline__ bf16x16 cat8(bf16x8 lo, bf16x8 hi) {
  return __builtin_shufflevector(lo, hi, 0,1,2,3,4,5,6,7,8,9,10,11,12,13,14,15);
}
__device__ __forceinline__ bf16x8 ld8(const bf16* p) {
  return *reinterpret_cast<const bf16x8*>(p);
}

// ---------------------------------------------------------------------------
// Generic bf16-WMMA GEMM body: C[m0:m0+128, n0:n0+64] = A*Bw + epilogue
// A: [M x K] (float or bf16), Bw: fp32 [K x N] row-major.
// C-frag: VGPR r -> row r + 8*(lane/16); col = lane%16.
// A-frag: elem e<8 -> k = g*8+e ; e>=8 -> k = 16+g*8+(e-8) (g = lane/16).
// B-frag: elem e<8 -> k = g*8+e ; e>=8 -> k = 16+g*8+(e-8).
// ---------------------------------------------------------------------------
template <typename AT>
__device__ __forceinline__ void gemm_body(
    const AT* __restrict__ A, int lda,
    const float* __restrict__ Bw, int ldb,
    const float* __restrict__ bias,
    const float* __restrict__ resid, int ldr,
    float* __restrict__ outF, bf16* __restrict__ outB, int ldo,
    int K, int m0, int n0, int act, int transOut)
{
  __shared__ alignas(16) bf16 As[BMT * ASTR];
  __shared__ alignas(16) bf16 Bt[BNT * ASTR];   // stored transposed: [n][k]

  const int t    = threadIdx.x;
  const int wave = t >> 5;
  const int lane = t & 31;
  const int lm   = lane & 15;
  const int lg   = lane >> 4;

  f32x8 acc[4];
  #pragma unroll
  for (int i = 0; i < 4; ++i)
    #pragma unroll
    for (int e = 0; e < 8; ++e) acc[i][e] = 0.0f;

  for (int k0 = 0; k0 < K; k0 += KCT) {
    __syncthreads();
    // ---- stage A tile: 128 x 32 -> bf16 LDS (wide loads, then stores) ----
    if constexpr (sizeof(AT) == 4) {
      float4 av[4];
      #pragma unroll
      for (int i = 0; i < 4; ++i) {             // 1024 float4 total
        int idx = t + i * WG;
        int r = idx >> 3, c4 = (idx & 7) * 4;
        av[i] = *(const float4*)&A[(size_t)(m0 + r) * lda + k0 + c4];
      }
      #pragma unroll
      for (int i = 0; i < 4; ++i) {
        int idx = t + i * WG;
        int r = idx >> 3, c4 = (idx & 7) * 4;
        bf16x4 o;
        o[0] = (bf16)av[i].x; o[1] = (bf16)av[i].y;
        o[2] = (bf16)av[i].z; o[3] = (bf16)av[i].w;
        *(bf16x4*)&As[r * ASTR + c4] = o;
      }
    } else {
      bf16x8 av[2];
      #pragma unroll
      for (int i = 0; i < 2; ++i) {             // 512 bf16x8 total
        int idx = t + i * WG;
        int r = idx >> 2, c8 = (idx & 3) * 8;
        av[i] = ld8(&A[(size_t)(m0 + r) * lda + k0 + c8]);
      }
      #pragma unroll
      for (int i = 0; i < 2; ++i) {
        int idx = t + i * WG;
        int r = idx >> 2, c8 = (idx & 3) * 8;
        *(bf16x8*)&As[r * ASTR + c8] = av[i];
      }
    }
    // ---- stage B tile transposed: Bt[n][k] (float4 loads, scatter) -------
    {
      float4 bl[2];
      #pragma unroll
      for (int i = 0; i < 2; ++i) {             // 512 float4 total
        int idx = t + i * WG;
        int kk = idx >> 4, n4 = (idx & 15) * 4;
        bl[i] = *(const float4*)&Bw[(size_t)(k0 + kk) * ldb + n0 + n4];
      }
      #pragma unroll
      for (int i = 0; i < 2; ++i) {
        int idx = t + i * WG;
        int kk = idx >> 4, n4 = (idx & 15) * 4;
        Bt[(n4 + 0) * ASTR + kk] = (bf16)bl[i].x;
        Bt[(n4 + 1) * ASTR + kk] = (bf16)bl[i].y;
        Bt[(n4 + 2) * ASTR + kk] = (bf16)bl[i].z;
        Bt[(n4 + 3) * ASTR + kk] = (bf16)bl[i].w;
      }
    }
    // prefetch next B tile rows (global_prefetch_b8 path)
    if (k0 + KCT < K)
      __builtin_prefetch(&Bw[(size_t)(k0 + KCT + (t >> 6)) * ldb + n0 + (t & 63)], 0, 3);
    __syncthreads();

    // ---- load all fragments first, then issue WMMAs back-to-back ---------
    const bf16* arow = &As[(wave * 16 + lm) * ASTR];
    bf16x16 afrag = cat8(ld8(arow + lg * 8), ld8(arow + 16 + lg * 8));

    bf16x16 bfr[4];
    #pragma unroll
    for (int nt = 0; nt < 4; ++nt) {
      const bf16* brow = &Bt[(nt * 16 + lm) * ASTR];
      bfr[nt] = cat8(ld8(brow + lg * 8), ld8(brow + 16 + lg * 8));
    }
    #pragma unroll
    for (int nt = 0; nt < 4; ++nt)
      acc[nt] = __builtin_amdgcn_wmma_f32_16x16x32_bf16(
          false, afrag, false, bfr[nt], (short)0, acc[nt], false, false);
  }

  // ---- epilogue ----------------------------------------------------------
  const int rowBase = m0 + wave * 16;
  #pragma unroll
  for (int nt = 0; nt < 4; ++nt) {
    const int col = n0 + nt * 16 + lm;
    const float bv = bias ? bias[col] : 0.0f;
    #pragma unroll
    for (int r = 0; r < 8; ++r) {
      const int row = rowBase + r + lg * 8;
      float v = acc[nt][r] + bv;
      if (resid) v += resid[(size_t)row * ldr + col];
      if (act == ACT_RELU) v = fmaxf(v, 0.0f);
      if (outF)          outF[(size_t)row * ldo + col] = v;
      else if (transOut) outB[(size_t)col * ldo + row] = (bf16)v;  // ldo = row stride of transposed out
      else               outB[(size_t)row * ldo + col] = (bf16)v;
    }
  }
}

// ---------------------------------------------------------------------------
// Kernels
// ---------------------------------------------------------------------------
__global__ __launch_bounds__(WG)
void gemm_af32(const float* A, int lda, const float* Bw, int ldb,
               const float* bias, const float* resid, int ldr,
               float* outF, bf16* outB, int ldo, int K, int act) {
  gemm_body<float>(A, lda, Bw, ldb, bias, resid, ldr, outF, outB, ldo,
                   K, blockIdx.x * BMT, blockIdx.y * BNT, act, 0);
}

__global__ __launch_bounds__(WG)
void gemm_abf16(const bf16* A, int lda, const float* Bw, int ldb,
                const float* bias, const float* resid, int ldr,
                float* outF, bf16* outB, int ldo, int K, int act) {
  gemm_body<bf16>(A, lda, Bw, ldb, bias, resid, ldr, outF, outB, ldo,
                  K, blockIdx.x * BMT, blockIdx.y * BNT, act, 0);
}

// Q/K/V projections: blockIdx.y in [0,48): mat = y/16 (Q,K,V), head = y%16.
// Q,K out bf16 [H][MT][DK]; V out TRANSPOSED bf16 [H][DK][MT].
__global__ __launch_bounds__(WG)
void qkv_kernel(const float* __restrict__ src,
                const float* __restrict__ Wq, const float* __restrict__ bq,
                const float* __restrict__ Wk, const float* __restrict__ bk,
                const float* __restrict__ Wv, const float* __restrict__ bv,
                bf16* __restrict__ qb, bf16* __restrict__ kb, bf16* __restrict__ vb) {
  const int mat = blockIdx.y >> 4;
  const int h   = blockIdx.y & 15;
  const float* W    = (mat == 0) ? Wq : (mat == 1) ? Wk : Wv;
  const float* bias = (mat == 0) ? bq : (mat == 1) ? bk : bv;
  bf16*        out  = (mat == 0) ? qb + (size_t)h * MT * DKK
                    : (mat == 1) ? kb + (size_t)h * MT * DKK
                                 : vb + (size_t)h * DKK * MT;
  const int ldo      = (mat == 2) ? MT : DKK;
  const int transOut = (mat == 2) ? 1 : 0;
  gemm_body<float>(src, DD, W + (size_t)h * DD * DKK, DKK, bias + h * DKK,
                   nullptr, 0, nullptr, out, ldo,
                   DD, blockIdx.x * BMT, 0, ACT_NONE, transOut);
}

// Flash-style attention: grid (SS/128, NB*HH). 8 waves x 16 query rows.
// K layout [H][MT][DK]; V layout [H][DK][MT] (pre-transposed).
__global__ __launch_bounds__(WG)
void attn_kernel(const bf16* __restrict__ qb, const bf16* __restrict__ kb,
                 const bf16* __restrict__ vb, float* __restrict__ attn_o) {
  constexpr int TK = 32;
  __shared__ alignas(16) bf16 Ks[TK * KSTR];        // [key][d]   (== B for QK^T)
  __shared__ alignas(16) bf16 Vt[DKK * ASTR];       // [dk][key]  (== B^T for PV)
  __shared__ alignas(16) bf16 Ps[8][16 * ASTR];     // per-wave P scratch

  const int t    = threadIdx.x;
  const int wave = t >> 5;
  const int lane = t & 31;
  const int lm   = lane & 15;
  const int lg   = lane >> 4;
  const int b    = blockIdx.y >> 4;
  const int h    = blockIdx.y & 15;

  const bf16* kbase  = kb + (size_t)h * MT * DKK + (size_t)b * SS * DKK; // [s][d]
  const bf16* vtbase = vb + (size_t)h * DKK * MT + (size_t)b * SS;       // [d][s], row stride MT
  const int qrow = b * SS + blockIdx.x * 128 + wave * 16;   // row into [MT]

  // per-thread staging coordinates (16B per lane per tile)
  const int skk = t >> 3, sd8 = (t & 7) * 8;   // K tile: 32 x 64
  const int sdk = t >> 2, sc8 = (t & 3) * 8;   // V tile: 64 x 32

  // Q fragments (two K=32 chunks over d), loaded once, straight from global
  const bf16* qrp = qb + (size_t)h * MT * DKK + (size_t)(qrow + lm) * DKK;
  bf16x16 aq[2];
  #pragma unroll
  for (int c = 0; c < 2; ++c)
    aq[c] = cat8(ld8(qrp + c * 32 + lg * 8), ld8(qrp + c * 32 + 16 + lg * 8));

  f32x8 oacc[4];
  #pragma unroll
  for (int i = 0; i < 4; ++i)
    #pragma unroll
    for (int e = 0; e < 8; ++e) oacc[i][e] = 0.0f;
  float mrun[8], lrun[8];
  #pragma unroll
  for (int r = 0; r < 8; ++r) { mrun[r] = -3.0e38f; lrun[r] = 0.0f; }

  for (int k0 = 0; k0 < SS; k0 += TK) {
    __syncthreads();
#if USE_ASYNC_LDS
    // ASYNCcnt-tracked direct global->LDS DMA (no VGPR round trip)
    ASYNC_CP16(kbase + (size_t)(k0 + skk) * DKK + sd8, &Ks[skk * KSTR + sd8]);
    ASYNC_CP16(vtbase + (size_t)sdk * MT + k0 + sc8,   &Vt[sdk * ASTR + sc8]);
    __builtin_amdgcn_s_wait_asynccnt(0);
#else
    *(bf16x8*)&Ks[skk * KSTR + sd8] = ld8(kbase + (size_t)(k0 + skk) * DKK + sd8);
    *(bf16x8*)&Vt[sdk * ASTR + sc8] = ld8(vtbase + (size_t)sdk * MT + k0 + sc8);
#endif
    __syncthreads();

    // ---- scores: load all 4 K-fragments, then 4 WMMAs back-to-back -------
    bf16x16 kf[2][2];
    #pragma unroll
    for (int st = 0; st < 2; ++st)
      #pragma unroll
      for (int c = 0; c < 2; ++c) {
        const bf16* kr = &Ks[(st * 16 + lm) * KSTR + c * 32];
        kf[st][c] = cat8(ld8(kr + lg * 8), ld8(kr + 16 + lg * 8));
      }
    f32x8 sv[2];
    #pragma unroll
    for (int st = 0; st < 2; ++st) {
      f32x8 s;
      #pragma unroll
      for (int e = 0; e < 8; ++e) s[e] = 0.0f;
      #pragma unroll
      for (int c = 0; c < 2; ++c)
        s = __builtin_amdgcn_wmma_f32_16x16x32_bf16(
                false, aq[c], false, kf[st][c], (short)0, s, false, false);
      #pragma unroll
      for (int e = 0; e < 8; ++e) s[e] *= 0.125f;   // 1/sqrt(DK)
      sv[st] = s;
    }

    // ---- online softmax (row-wise; 16-lane groups == 16 key columns) -----
    float mnew[8], alpha[8];
    #pragma unroll
    for (int r = 0; r < 8; ++r) {
      float a = fmaxf(sv[0][r], sv[1][r]);
      #pragma unroll
      for (int off = 8; off; off >>= 1) a = fmaxf(a, __shfl_xor(a, off, 16));
      float mn = fmaxf(mrun[r], a);
      mnew[r] = mn;
      alpha[r] = __expf(mrun[r] - mn);
      mrun[r] = mn;
    }
    #pragma unroll
    for (int r = 0; r < 8; ++r) {
      float p0 = __expf(sv[0][r] - mnew[r]);
      float p1 = __expf(sv[1][r] - mnew[r]);
      sv[0][r] = p0; sv[1][r] = p1;
      float sum = p0 + p1;
      #pragma unroll
      for (int off = 8; off; off >>= 1) sum += __shfl_xor(sum, off, 16);
      lrun[r] = lrun[r] * alpha[r] + sum;
    }
    #pragma unroll
    for (int nt = 0; nt < 4; ++nt)
      #pragma unroll
      for (int r = 0; r < 8; ++r) oacc[nt][r] *= alpha[r];

    // ---- reshape P: C-frag -> LDS -> A-frag ------------------------------
    bf16* pw = &Ps[wave][0];
    #pragma unroll
    for (int st = 0; st < 2; ++st)
      #pragma unroll
      for (int r = 0; r < 8; ++r)
        pw[(r + lg * 8) * ASTR + st * 16 + lm] = (bf16)sv[st][r];
    __syncthreads();

    const bf16* pr = pw + lm * ASTR;
    bf16x16 pf = cat8(ld8(pr + lg * 8), ld8(pr + 16 + lg * 8));

    // ---- O += P @ V: load all 4 V-fragments, then 4 WMMAs ----------------
    bf16x16 vf[4];
    #pragma unroll
    for (int nt = 0; nt < 4; ++nt) {
      const bf16* vr = &Vt[(nt * 16 + lm) * ASTR];
      vf[nt] = cat8(ld8(vr + lg * 8), ld8(vr + 16 + lg * 8));
    }
    #pragma unroll
    for (int nt = 0; nt < 4; ++nt)
      oacc[nt] = __builtin_amdgcn_wmma_f32_16x16x32_bf16(
          false, pf, false, vf[nt], (short)0, oacc[nt], false, false);
  }

  // ---- write O / l to [MT][D] at columns h*64.. --------------------------
  #pragma unroll
  for (int nt = 0; nt < 4; ++nt) {
    const int col = h * DKK + nt * 16 + lm;
    #pragma unroll
    for (int r = 0; r < 8; ++r) {
      const int row = qrow + r + lg * 8;
      attn_o[(size_t)row * DD + col] = oacc[nt][r] / lrun[r];
    }
  }
}

// Row LayerNorm: one block per row of 1024, float4 I/O.
__global__ __launch_bounds__(WG)
void ln_kernel(const float* __restrict__ in, const float* __restrict__ gam,
               const float* __restrict__ bet, float* __restrict__ out) {
  __shared__ float rs[8], rq[8];
  const int row = blockIdx.x;
  const int t = threadIdx.x;
  const float4 v = ((const float4*)(in + (size_t)row * DD))[t];
  float s = v.x + v.y + v.z + v.w;
  float q = v.x * v.x + v.y * v.y + v.z * v.z + v.w * v.w;
  #pragma unroll
  for (int off = 16; off; off >>= 1) {
    s += __shfl_xor(s, off, 32);
    q += __shfl_xor(q, off, 32);
  }
  if ((t & 31) == 0) { rs[t >> 5] = s; rq[t >> 5] = q; }
  __syncthreads();
  float ts = 0.0f, tq = 0.0f;
  #pragma unroll
  for (int w = 0; w < 8; ++w) { ts += rs[w]; tq += rq[w]; }
  const float mean = ts * (1.0f / DD);
  const float var  = tq * (1.0f / DD) - mean * mean;
  const float rstd = rsqrtf(var + 1e-5f);
  const float4 g4 = ((const float4*)gam)[t];
  const float4 b4 = ((const float4*)bet)[t];
  float4 o;
  o.x = (v.x - mean) * rstd * g4.x + b4.x;
  o.y = (v.y - mean) * rstd * g4.y + b4.y;
  o.z = (v.z - mean) * rstd * g4.z + b4.z;
  o.w = (v.w - mean) * rstd * g4.w + b4.w;
  ((float4*)(out + (size_t)row * DD))[t] = o;
}

// ---------------------------------------------------------------------------
// Launcher
// ---------------------------------------------------------------------------
extern "C" void kernel_launch(void* const* d_in, const int* in_sizes, int n_in,
                              void* d_out, int out_size, void* d_ws, size_t ws_size,
                              hipStream_t stream) {
  const float* src = (const float*)d_in[0];
  const float* Wq  = (const float*)d_in[1];
  const float* bq  = (const float*)d_in[2];
  const float* Wk  = (const float*)d_in[3];
  const float* bk  = (const float*)d_in[4];
  const float* Wv  = (const float*)d_in[5];
  const float* bv  = (const float*)d_in[6];
  const float* Wo  = (const float*)d_in[7];
  const float* bo  = (const float*)d_in[8];
  const float* g1  = (const float*)d_in[9];
  const float* be1 = (const float*)d_in[10];
  const float* W1  = (const float*)d_in[11];
  const float* b1  = (const float*)d_in[12];
  const float* W2  = (const float*)d_in[13];
  const float* b2  = (const float*)d_in[14];
  const float* g2  = (const float*)d_in[15];
  const float* be2 = (const float*)d_in[16];
  float* out = (float*)d_out;

  // workspace carve-up
  char* p = (char*)d_ws;
  const size_t qkvB = (size_t)HH * MT * DKK * sizeof(bf16);  // 8 MB each
  const size_t matB = (size_t)MT * DD * sizeof(float);       // 16 MB each
  bf16*  qb     = (bf16*)p;  p += qkvB;
  bf16*  kbuf   = (bf16*)p;  p += qkvB;
  bf16*  vbuf   = (bf16*)p;  p += qkvB;    // transposed [H][DK][MT]
  float* attn_o = (float*)p; p += matB;
  float* tmp1   = (float*)p; p += matB;
  float* xbuf   = (float*)p; p += matB;
  float* tmp2   = (float*)p; p += matB;
  bf16*  h1     = (bf16*)p;  // MT*DFF bf16 = 32 MB

  // 1) Q/K/V projections (bf16 out; V written transposed)
  qkv_kernel<<<dim3(MT / BMT, 48), WG, 0, stream>>>(
      src, Wq, bq, Wk, bk, Wv, bv, qb, kbuf, vbuf);

  // 2) attention -> attn_o [MT][D] (heads concatenated)
  attn_kernel<<<dim3(SS / 128, NB * HH), WG, 0, stream>>>(qb, kbuf, vbuf, attn_o);

  // 3) out-proj + bias + residual(src) -> tmp1
  gemm_af32<<<dim3(MT / BMT, DD / BNT), WG, 0, stream>>>(
      attn_o, DD, Wo, DD, bo, src, DD, tmp1, nullptr, DD, DD, ACT_NONE);

  // 4) layernorm1 -> x
  ln_kernel<<<MT, WG, 0, stream>>>(tmp1, g1, be1, xbuf);

  // 5) FFN1: relu(x@W1 + b1) -> h1 (bf16)
  gemm_af32<<<dim3(MT / BMT, DFF / BNT), WG, 0, stream>>>(
      xbuf, DD, W1, DFF, b1, nullptr, 0, nullptr, h1, DFF, DD, ACT_RELU);

  // 6) FFN2: h1@W2 + b2 + residual(x) -> tmp2
  gemm_abf16<<<dim3(MT / BMT, DD / BNT), WG, 0, stream>>>(
      h1, DFF, W2, DD, b2, xbuf, DD, tmp2, nullptr, DD, DFF, ACT_NONE);

  // 7) layernorm2 -> output
  ln_kernel<<<MT, WG, 0, stream>>>(tmp2, g2, be2, out);
}
